// ks_net_block1_16810501996766
// MI455X (gfx1250) — matrix operand order repaired and checked by reading, not compiled, not run on hardware
//
#include <hip/hip_runtime.h>
#include <math.h>

#define HWD 128
#define LPIX (HWD * HWD)

typedef __attribute__((ext_vector_type(16))) _Float16 v16h;
typedef __attribute__((ext_vector_type(8)))  float    v8f;

__device__ __forceinline__ float siluf(float x)    { return x / (1.f + __expf(-x)); }
__device__ __forceinline__ float sigmf(float x)    { return 1.f / (1.f + __expf(-x)); }
__device__ __forceinline__ float softplusf_(float x){ return (x > 30.f) ? x : log1pf(__expf(x)); }

// -------------------------------------------------------------------------
// Guard-free WMMA GEMM: C(MT*16 x N) = A(MT*16 x 32) * B(32 x N)
//   A: row-major, lda = 32 (compile-time) -> 4x global_load_b128 per fragment
//   B: row stride sb_r, col stride 1      -> 16 coalesced scalar loads
//   C: row stride sc_r, col stride 1
// One wave32 per 16-wide N tile; MT stacked M tiles share the B fragment.
// Fragment packing per CDNA5 ISA 7.12.2 (f16 A 16x32 / B 32x16 / f32 C 16x16).
// -------------------------------------------------------------------------
template<int MT>
__global__ void gemm32_wmma_k(const float* __restrict__ A,
                              const float* __restrict__ B, long sb_r,
                              float* __restrict__ C, long sc_r)
{
  const int lane = threadIdx.x & 31;
  const int hf   = lane >> 4;       // lane half selects K sub-range
  const int l16  = lane & 15;
  const int n0   = blockIdx.x * 16;
  const int mb   = blockIdx.y * (MT * 16);

  // B fragment: lane = column n, idx i -> k = 16*(i/8) + 8*half + i%8
  v16h b;
  const float* bp = B + (long)(hf * 8) * sb_r + (n0 + l16);
#pragma unroll
  for (int i = 0; i < 16; ++i) b[i] = (_Float16)bp[(long)((i & 7) + ((i >> 3) << 4)) * sb_r];

  // A fragments: lane row = m, idx i -> k = 16*(i/8) + 8*half + i%8
  v16h a[MT];
#pragma unroll
  for (int t = 0; t < MT; ++t) {
    const float* ap = A + (long)(mb + t * 16 + l16) * 32 + hf * 8;
    float tmp[16];
    *(float4*)(tmp + 0)  = *(const float4*)(ap + 0);
    *(float4*)(tmp + 4)  = *(const float4*)(ap + 4);
    *(float4*)(tmp + 8)  = *(const float4*)(ap + 16);
    *(float4*)(tmp + 12) = *(const float4*)(ap + 20);
#pragma unroll
    for (int i = 0; i < 16; ++i) a[t][i] = (_Float16)tmp[i];
  }

  v8f acc[MT];
#pragma unroll
  for (int t = 0; t < MT; ++t) {
    v8f z = {0.f, 0.f, 0.f, 0.f, 0.f, 0.f, 0.f, 0.f};
    acc[t] = __builtin_amdgcn_wmma_f32_16x16x32_f16(false, a[t], false, b,
                                                    (short)0, z, false, false);
  }

  // C/D layout: VGPR r -> M = r + 8*half, N = lane%16
#pragma unroll
  for (int t = 0; t < MT; ++t)
#pragma unroll
    for (int r = 0; r < 8; ++r)
      C[(long)(mb + t * 16 + (hf << 3) + r) * sc_r + (n0 + l16)] = acc[t][r];
}

// Zero-pad rows: dst(Rdst x Ccols) <- src(Rsrc x Ccols), rows >= Rsrc zeroed
__global__ void pad_rows_k(const float* __restrict__ src, float* __restrict__ dst,
                           int Rsrc, int Rdst, int Ccols)
{
  int idx = blockIdx.x * blockDim.x + threadIdx.x;
  if (idx >= Rdst * Ccols) return;
  int r = idx / Ccols;
  dst[idx] = (r < Rsrc) ? src[idx] : 0.f;
}

// -------------------------------------------------------------------------
// Generic 3x3 SAME conv, strided in/out layouts (handles NCHW and NHWC).
// act: 0=none 1=gelu(exact) 2=leaky_relu(0.01) 3=abs
// -------------------------------------------------------------------------
__global__ void conv3x3_k(const float* __restrict__ in, int Cin, long in_l, long in_c,
                          const float* __restrict__ w, const float* __restrict__ bias,
                          const float* __restrict__ res,
                          float* __restrict__ out, int Cout, long out_l, long out_c,
                          int act)
{
  long idx = (long)blockIdx.x * blockDim.x + threadIdx.x;
  if (idx >= (long)LPIX * Cout) return;
  int co = (int)(idx % Cout);
  int l  = (int)(idx / Cout);
  int px = l % HWD, py = l / HWD;
  float s = bias ? bias[co] : 0.f;
  for (int ky = -1; ky <= 1; ++ky) {
    int yy = py + ky; if (yy < 0 || yy >= HWD) continue;
    for (int kx = -1; kx <= 1; ++kx) {
      int xx = px + kx; if (xx < 0 || xx >= HWD) continue;
      long ip = (long)(yy * HWD + xx);
      const float* wp = w + (long)co * Cin * 9 + (ky + 1) * 3 + (kx + 1);
      for (int ci = 0; ci < Cin; ++ci)
        s += in[ip * in_l + (long)ci * in_c] * wp[(long)ci * 9];
    }
  }
  if      (act == 1) s = 0.5f * s * (1.f + erff(s * 0.70710678118654752f));
  else if (act == 2) s = (s > 0.f) ? s : 0.01f * s;
  else if (act == 3) s = fabsf(s);
  if (res) s += res[(long)l * Cout + co];
  out[(long)l * out_l + (long)co * out_c] = s;
}

// LayerNorm over C channels of (L x C); output row stride Cout, cols [C,Cout) zeroed
__global__ void ln_k(const float* __restrict__ in, const float* __restrict__ g,
                     const float* __restrict__ b, float* __restrict__ out,
                     int C, int Cout)
{
  int l = blockIdx.x * blockDim.x + threadIdx.x;
  if (l >= LPIX) return;
  const float* p = in + (long)l * C;
  float m = 0.f;
  for (int c = 0; c < C; ++c) m += p[c];
  m /= (float)C;
  float v = 0.f;
  for (int c = 0; c < C; ++c) { float d = p[c] - m; v += d * d; }
  v /= (float)C;
  float r = rsqrtf(v + 1e-5f);
  float* q = out + (long)l * Cout;
  for (int c = 0; c < C; ++c) q[c] = (p[c] - m) * r * g[c] + b[c];
  for (int c = C; c < Cout; ++c) q[c] = 0.f;
}

// Depthwise 3x3 conv + bias + SiLU; input = xz columns [0,32) in (L x 64),
// output channel-major (32 x L)
__global__ void dwconv_silu_k(const float* __restrict__ xz, const float* __restrict__ w,
                              const float* __restrict__ b, float* __restrict__ xc)
{
  long idx = (long)blockIdx.x * blockDim.x + threadIdx.x;
  if (idx >= 32L * LPIX) return;
  int d = (int)(idx / LPIX);
  int l = (int)(idx % LPIX);
  int px = l % HWD, py = l / HWD;
  float s = b[d];
  for (int ky = -1; ky <= 1; ++ky) {
    int yy = py + ky; if (yy < 0 || yy >= HWD) continue;
    for (int kx = -1; kx <= 1; ++kx) {
      int xx = px + kx; if (xx < 0 || xx >= HWD) continue;
      s += xz[(long)(yy * HWD + xx) * 64 + d] * w[d * 9 + (ky + 1) * 3 + (kx + 1)];
    }
  }
  xc[(long)d * LPIX + l] = siluf(s);
}

// Build the 4 cross-scan directions: row, col, and both reversed
__global__ void scatter_xs_k(const float* __restrict__ xc, float* __restrict__ xs)
{
  long idx = (long)blockIdx.x * blockDim.x + threadIdx.x;
  if (idx >= 32L * LPIX) return;
  int d = (int)(idx / LPIX);
  int l = (int)(idx % LPIX);
  float v = xc[(long)d * LPIX + l];
  int py = l / HWD, px = l % HWD;
  int lt = px * HWD + py;
  const long DL = 32L * LPIX;
  xs[0 * DL + (long)d * LPIX + l]               = v;
  xs[1 * DL + (long)d * LPIX + lt]              = v;
  xs[2 * DL + (long)d * LPIX + (LPIX - 1 - l)]  = v;
  xs[3 * DL + (long)d * LPIX + (LPIX - 1 - lt)] = v;
}

// Selective scan: one 16-lane group per (direction k, channel d), lane = state n.
// xdbl layout is channel-major: row c (0..47) x column l, row stride L.
// h[l] = exp(dt*A_n) * h[l-1] + dt*x*B_n ; y[l] = sum_n h[l]*C_n + D*x
__global__ void scan_k(const float* __restrict__ xs, const float* __restrict__ xdbl,
                       const float* __restrict__ dt_w, const float* __restrict__ dt_b,
                       const float* __restrict__ A_log, const float* __restrict__ Dp,
                       float* __restrict__ y)
{
  int gid = blockIdx.x * blockDim.x + threadIdx.x;
  int grp = gid >> 4;              // (k,d) pair, 0..127
  int n   = gid & 15;              // state index
  if (grp >= 128) return;
  int k = grp >> 5, d = grp & 31;
  float dtw = dt_w[k * 32 + d];
  float dtb = dt_b[k * 32 + d];
  float An  = -__expf(A_log[(k * 32 + d) * 16 + n]);
  float Dv  = Dp[k * 32 + d];
  const float* xsp = xs + ((long)k * 32 + d) * LPIX;
  const float* dtp = xdbl + (long)k * 48 * LPIX;        // row 0: dt_rank scalar
  const float* Bp  = dtp + (long)(1 + n) * LPIX;        // rows 1..16: B states
  const float* Cp  = dtp + (long)(17 + n) * LPIX;       // rows 17..32: C states
  float* yp = y + ((long)k * 32 + d) * LPIX;
  float hst = 0.f;
  for (int l = 0; l < LPIX; ++l) {
    __builtin_prefetch(xsp + l + 64, 0, 1);             // -> global_prefetch_b8
    float xv  = xsp[l];
    float dt  = softplusf_(dtw * dtp[l] + dtb);
    float dA  = __expf(dt * An);
    hst = dA * hst + dt * xv * Bp[l];
    float c = hst * Cp[l];
    c += __shfl_xor(c, 8, 16);
    c += __shfl_xor(c, 4, 16);
    c += __shfl_xor(c, 2, 16);
    c += __shfl_xor(c, 1, 16);
    if (n == 0) yp[l] = c + Dv * xv;
  }
}

// Merge 4 direction outputs (with flips / transposes), out-LN over 32 ch,
// gate with silu(z) from xz columns [32,64). Output (L x 32) row-major.
__global__ void merge_ln_gate_k(const float* __restrict__ y, const float* __restrict__ xz,
                                const float* __restrict__ g, const float* __restrict__ b,
                                float* __restrict__ yt)
{
  int l = blockIdx.x * blockDim.x + threadIdx.x;
  if (l >= LPIX) return;
  int py = l / HWD, px = l % HWD;
  int lt = px * HWD + py;
  const long DL = 32L * LPIX;
  float v[32];
  float m = 0.f;
  for (int d = 0; d < 32; ++d) {
    long dl = (long)d * LPIX;
    float t = y[0 * DL + dl + l]
            + y[2 * DL + dl + (LPIX - 1 - l)]
            + y[1 * DL + dl + lt]
            + y[3 * DL + dl + (LPIX - 1 - lt)];
    v[d] = t; m += t;
  }
  m *= (1.f / 32.f);
  float var = 0.f;
  for (int d = 0; d < 32; ++d) { float t = v[d] - m; var += t * t; }
  var *= (1.f / 32.f);
  float r = rsqrtf(var + 1e-5f);
  for (int d = 0; d < 32; ++d) {
    float z = xz[(long)l * 64 + 32 + d];
    yt[(long)l * 32 + d] = ((v[d] - m) * r * g[d] + b[d]) * siluf(z);
  }
}

// x = x * scale[c] + br * (att ? att[c] : 1)
__global__ void scale_add_k(float* __restrict__ x, const float* __restrict__ scale,
                            const float* __restrict__ br, const float* __restrict__ att,
                            int C)
{
  long idx = (long)blockIdx.x * blockDim.x + threadIdx.x;
  if (idx >= (long)LPIX * C) return;
  int c = (int)(idx % C);
  float gt = att ? att[c] : 1.f;
  x[idx] = x[idx] * scale[c] + br[idx] * gt;
}

// Per-channel spatial mean of (L x 16)
__global__ void cab_mean_k(const float* __restrict__ h2, float* __restrict__ amean)
{
  __shared__ float sm[256];
  int c = blockIdx.x;
  float s = 0.f;
  for (int l = threadIdx.x; l < LPIX; l += 256) s += h2[(long)l * 16 + c];
  sm[threadIdx.x] = s;
  __syncthreads();
  for (int o = 128; o > 0; o >>= 1) {
    if ((int)threadIdx.x < o) sm[threadIdx.x] += sm[threadIdx.x + o];
    __syncthreads();
  }
  if (threadIdx.x == 0) amean[c] = sm[0] / (float)LPIX;
}

// Channel attention squeeze(16->1)->relu->excite(1->16)->sigmoid
__global__ void cab_att_k(const float* __restrict__ amean,
                          const float* __restrict__ a1w, const float* __restrict__ a1b,
                          const float* __restrict__ a2w, const float* __restrict__ a2b,
                          float* __restrict__ att)
{
  int lane = threadIdx.x;
  float v = (lane < 16) ? a1w[lane] * amean[lane] : 0.f;
  v += __shfl_xor(v, 16);
  v += __shfl_xor(v, 8);
  v += __shfl_xor(v, 4);
  v += __shfl_xor(v, 2);
  v += __shfl_xor(v, 1);
  float s = fmaxf(v + a1b[0], 0.f);
  if (lane < 16) att[lane] = sigmf(a2w[lane] * s + a2b[lane]);
}

// -------------------------------------------------------------------------
// Host orchestration
// -------------------------------------------------------------------------
struct SS2DP {
  const float *in_proj, *conv_w, *conv_b, *x_proj, *dt_w, *dt_b, *A_log, *Dp,
              *out_ln_g, *out_ln_b, *out_proj;
};
struct CABP { const float *c1w, *c1b, *c2w, *c2b, *a1w, *a1b, *a2w, *a2b; };
struct BlockP {
  const float *ln1g, *ln1b, *scale1;
  SS2DP ss;
  const float *ln2g, *ln2b, *scale2;
  CABP cab;
};

extern "C" void kernel_launch(void* const* d_in, const int* in_sizes, int n_in,
                              void* d_out, int out_size, void* d_ws, size_t ws_size,
                              hipStream_t stream)
{
  (void)in_sizes; (void)n_in; (void)out_size; (void)ws_size;
  int ii = 0;
  auto nxt = [&]() { return (const float*)d_in[ii++]; };

  // setup_inputs() dict insertion order: x, then params
  const float* x   = nxt();
  const float* cfw = nxt(); const float* cfb = nxt();
  const float* peg = nxt(); const float* peb = nxt();
  const int depths[2] = {1, 2};
  BlockP blocks[3];
  const float* gconvw[2]; const float* gconvb[2];
  int bi = 0;
  for (int g = 0; g < 2; ++g) {
    for (int dI = 0; dI < depths[g]; ++dI) {
      BlockP& B = blocks[bi++];
      B.ln1g = nxt(); B.ln1b = nxt(); B.scale1 = nxt();
      B.ss.in_proj = nxt(); B.ss.conv_w = nxt(); B.ss.conv_b = nxt();
      B.ss.x_proj = nxt(); B.ss.dt_w = nxt(); B.ss.dt_b = nxt();
      B.ss.A_log = nxt(); B.ss.Dp = nxt();
      B.ss.out_ln_g = nxt(); B.ss.out_ln_b = nxt(); B.ss.out_proj = nxt();
      B.ln2g = nxt(); B.ln2b = nxt(); B.scale2 = nxt();
      B.cab.c1w = nxt(); B.cab.c1b = nxt(); B.cab.c2w = nxt(); B.cab.c2b = nxt();
      B.cab.a1w = nxt(); B.cab.a1b = nxt(); B.cab.a2w = nxt(); B.cab.a2b = nxt();
    }
    gconvw[g] = nxt(); gconvb[g] = nxt();
  }
  const float* ng  = nxt(); const float* nb  = nxt();
  const float* caw = nxt(); const float* cabw = nxt();
  const float* cuw = nxt(); const float* cub = nxt();
  const float* clw = nxt(); const float* clb = nxt();

  // workspace carve (f32)
  float* wsp = (float*)d_ws;
  auto carve = [&](size_t nf) { float* p = wsp; wsp += nf; return p; };
  const size_t Ls = (size_t)LPIX;
  float* f      = carve(Ls * 16);     // conv_first output, NHWC (L x 16)
  float* h      = carve(Ls * 16);     // running feature map (L x 16)
  float* resb   = carve(Ls * 16);     // group residual
  float* xlnb   = carve(Ls * 32);     // LN output, padded to 32 cols for WMMA
  float* xz     = carve(Ls * 64);     // in_proj output
  float* xc     = carve(32 * Ls);     // depthwise-conv output (32 x L)
  float* xs     = carve(4 * 32 * Ls); // 4 scan directions
  float* xdbl   = carve(4 * 48 * Ls); // channel-major: [dt | B(16) | C(16) | pad] x L
  float* ybuf   = carve(4 * 32 * Ls); // scan outputs
  float* yt     = carve(Ls * 32);     // merged + LN + gated
  float* sso    = carve(Ls * 16);     // out_proj output
  float* c1buf  = carve(Ls * 8);      // CAB hidden (5 used)
  float* h2     = carve(Ls * 16);     // CAB conv output
  float* amean  = carve(16);
  float* att    = carve(16);
  float* t1     = carve(Ls * 16);
  float* t2     = carve(Ls * 64);
  float* ipw    = carve(32 * 64);     // in_proj padded to 32 rows
  float* xpw    = carve(4 * 48 * 32); // x_proj padded to 48 rows per direction

  dim3 B256(256);
  auto g1 = [](long n) { return dim3((unsigned)((n + 255) / 256)); };

  // conv_first: NCHW input (17 x L) -> NHWC f (L x 16)
  conv3x3_k<<<g1((long)LPIX * 16), B256, 0, stream>>>(
      x, 17, 1, LPIX, cfw, cfb, nullptr, f, 16, 16, 1, 0);
  // patch-embed LN
  ln_k<<<g1(LPIX), B256, 0, stream>>>(f, peg, peb, h, 16, 16);

  bi = 0;
  for (int g = 0; g < 2; ++g) {
    hipMemcpyAsync(resb, h, Ls * 16 * sizeof(float), hipMemcpyDeviceToDevice, stream);
    for (int dI = 0; dI < depths[g]; ++dI) {
      const BlockP& Bp = blocks[bi++];
      // ---------- SS2D branch ----------
      // weight padding (K padded to 32 / M padded to 48 -> guard-free WMMA)
      pad_rows_k<<<g1(32 * 64), B256, 0, stream>>>(Bp.ss.in_proj, ipw, 16, 32, 64);
      for (int k = 0; k < 4; ++k)
        pad_rows_k<<<g1(48 * 32), B256, 0, stream>>>(
            Bp.ss.x_proj + (size_t)k * 33 * 32, xpw + (size_t)k * 48 * 32, 33, 48, 32);

      ln_k<<<g1(LPIX), B256, 0, stream>>>(h, Bp.ln1g, Bp.ln1b, xlnb, 16, 32);
      // xz(L x 64) = xlnb(L x 32, zero-padded) @ ipw(32 x 64)   [WMMA]
      gemm32_wmma_k<1><<<dim3(4, LPIX / 16), dim3(32), 0, stream>>>(
          xlnb, ipw, 64, xz, 64);
      dwconv_silu_k<<<g1(32L * LPIX), B256, 0, stream>>>(xz, Bp.ss.conv_w, Bp.ss.conv_b, xc);
      scatter_xs_k<<<g1(32L * LPIX), B256, 0, stream>>>(xc, xs);
      // xdbl_k(48 x L) = xpw_k(48 x 32) @ xs_k(32 x L)   [WMMA, 3 M-tiles/wave]
      for (int k = 0; k < 4; ++k)
        gemm32_wmma_k<3><<<dim3(LPIX / 16, 1), dim3(32), 0, stream>>>(
            xpw + (size_t)k * 48 * 32, xs + (size_t)k * 32 * Ls, LPIX,
            xdbl + (size_t)k * 48 * Ls, LPIX);
      scan_k<<<dim3(8), B256, 0, stream>>>(
          xs, xdbl, Bp.ss.dt_w, Bp.ss.dt_b, Bp.ss.A_log, Bp.ss.Dp, ybuf);
      merge_ln_gate_k<<<g1(LPIX), B256, 0, stream>>>(
          ybuf, xz, Bp.ss.out_ln_g, Bp.ss.out_ln_b, yt);
      // sso(L x 16) = yt(L x 32) @ out_proj(32 x 16)   [WMMA]
      gemm32_wmma_k<1><<<dim3(1, LPIX / 16), dim3(32), 0, stream>>>(
          yt, Bp.ss.out_proj, 16, sso, 16);
      scale_add_k<<<g1((long)LPIX * 16), B256, 0, stream>>>(h, Bp.scale1, sso, nullptr, 16);
      // ---------- CAB branch ----------
      ln_k<<<g1(LPIX), B256, 0, stream>>>(h, Bp.ln2g, Bp.ln2b, xlnb, 16, 16);
      conv3x3_k<<<g1((long)LPIX * 5), B256, 0, stream>>>(
          xlnb, 16, 16, 1, Bp.cab.c1w, Bp.cab.c1b, nullptr, c1buf, 5, 5, 1, 1);
      conv3x3_k<<<g1((long)LPIX * 16), B256, 0, stream>>>(
          c1buf, 5, 5, 1, Bp.cab.c2w, Bp.cab.c2b, nullptr, h2, 16, 16, 1, 0);
      cab_mean_k<<<dim3(16), B256, 0, stream>>>(h2, amean);
      cab_att_k<<<dim3(1), dim3(32), 0, stream>>>(
          amean, Bp.cab.a1w, Bp.cab.a1b, Bp.cab.a2w, Bp.cab.a2b, att);
      scale_add_k<<<g1((long)LPIX * 16), B256, 0, stream>>>(h, Bp.scale2, h2, att, 16);
    }
    // group tail: h = resb + conv3x3(h)
    conv3x3_k<<<g1((long)LPIX * 16), B256, 0, stream>>>(
        h, 16, 16, 1, gconvw[g], gconvb[g], resb, t1, 16, 16, 1, 0);
    hipMemcpyAsync(h, t1, Ls * 16 * sizeof(float), hipMemcpyDeviceToDevice, stream);
  }

  // tail: norm -> conv_after + f -> conv_up + lrelu -> conv_last -> abs (NCHW out)
  ln_k<<<g1(LPIX), B256, 0, stream>>>(h, ng, nb, xlnb, 16, 16);
  conv3x3_k<<<g1((long)LPIX * 16), B256, 0, stream>>>(
      xlnb, 16, 16, 1, caw, cabw, f, t1, 16, 16, 1, 0);
  conv3x3_k<<<g1((long)LPIX * 64), B256, 0, stream>>>(
      t1, 16, 16, 1, cuw, cub, nullptr, t2, 64, 64, 1, 2);
  conv3x3_k<<<g1((long)LPIX * 17), B256, 0, stream>>>(
      t2, 64, 64, 1, clw, clb, nullptr, (float*)d_out, 17, 1, LPIX, 3);
}
